// SpatioTemporalEncoder_4612794876553
// MI455X (gfx1250) — compile-verified, matrix operand order checked by reading, not run on hardware
//
#include <hip/hip_runtime.h>

typedef __attribute__((ext_vector_type(16))) _Float16 v16h;
typedef __attribute__((ext_vector_type(8)))  float    v8f;

#define T_DIM   100
#define MROWS   200            // T*C rows of the flattened hist
#define MPAD    208            // 13 * 16
#define MT      13             // M tiles of 16
#define KDIM    16384          // H*W
#define DDIM    8192
#define BN      128            // output columns per workgroup
#define KSPLIT  4
#define KCHUNK  (KDIM / KSPLIT)  // 4096
#define KK      32             // WMMA K step
#define LDA     34             // padded halfs per A row  (bank-conflict pad)
#define LDB     34             // padded halfs per B column (transposed tile)

union Frag16 { v16h v; uint32_t u[8]; };

__global__ __launch_bounds__(256) void hv_gemm_bind_kernel(
    const float* __restrict__ hist,    // [200, 16384]
    const float* __restrict__ time_w,  // [100, 8192]
    const float* __restrict__ pol_w,   // [2, 8192]
    const float* __restrict__ pos_w,   // [16384, 8192]
    float* __restrict__ gesture)       // [8192] accumulator (pre-zeroed)
{
    __shared__ _Float16 aTile[MPAD * LDA];   // hist tile, row-major [row][k]
    __shared__ _Float16 bTile[BN * LDB];     // pos_w tile transposed [n][k]

    const int tid     = threadIdx.x;
    const int lane    = tid & 31;
    const int wave    = tid >> 5;            // 0..7, 16 columns each
    const int colBase = blockIdx.x * BN;
    const int k0base  = blockIdx.y * KCHUNK;

    v8f acc[MT];
    #pragma unroll
    for (int m = 0; m < MT; ++m) acc[m] = (v8f)0.f;

    const int halfSel = lane >> 4;           // 0: lanes 0-15, 1: lanes 16-31
    const int l16     = lane & 15;
    const int kbase   = halfSel * 8;         // K sub-block per half-wave (ISA layout)

    for (int ks = 0; ks < KCHUNK; ks += KK) {
        const int k0 = k0base + ks;

        // ---- stage A: hist rows 0..207 (zero-padded past 199), 32 K values
        for (int i = tid; i < MPAD * KK; i += 256) {
            int r = i >> 5;                  // 0..207
            int k = i & 31;
            float v = (r < MROWS) ? hist[(size_t)r * KDIM + (k0 + k)] : 0.f;
            aTile[r * LDA + k] = (_Float16)v;
        }
        // ---- stage B transposed: bTile[n][k] = pos_w[k0+k][colBase+n]
        for (int i = tid; i < BN * KK; i += 256) {
            int n = i & (BN - 1);            // column fastest -> coalesced global
            int k = i >> 7;                  // 0..31
            float v = pos_w[(size_t)(k0 + k) * DDIM + (colBase + n)];
            bTile[n * LDB + k] = (_Float16)v;
        }
        __syncthreads();

        // ---- build B fragment for this wave's 16 columns
        // 16-bit operand layout: VGPR g<4 holds K = kbase+2g,2g+1 ; g>=4 holds 16+...
        Frag16 bf;
        {
            const int n = wave * 16 + l16;
            const uint32_t* bU = (const uint32_t*)&bTile[n * LDB];
            #pragma unroll
            for (int g = 0; g < 8; ++g) {
                int K = ((g & 4) ? 16 : 0) + kbase + ((g & 3) << 1);
                bf.u[g] = bU[K >> 1];
            }
        }

        // ---- 13 WMMAs: all M tiles against this wave's B fragment
        #pragma unroll
        for (int m = 0; m < MT; ++m) {
            Frag16 af;
            const int row = m * 16 + l16;
            const uint32_t* aU = (const uint32_t*)&aTile[row * LDA];
            #pragma unroll
            for (int g = 0; g < 8; ++g) {
                int K = ((g & 4) ? 16 : 0) + kbase + ((g & 3) << 1);
                af.u[g] = aU[K >> 1];
            }
            acc[m] = __builtin_amdgcn_wmma_f32_16x16x32_f16(
                false, af.v, false, bf.v, (short)0, acc[m], false, false);
        }
        __syncthreads();
    }

    // ---- bind with time/polarity hypervectors, reduce over rows, accumulate
    // C/D layout: VGPR r -> row m*16 + r + 8*halfSel, lane%16 -> column
    const int d = colBase + wave * 16 + l16;
    float partial = 0.f;
    #pragma unroll
    for (int m = 0; m < MT; ++m) {
        #pragma unroll
        for (int r = 0; r < 8; ++r) {
            int M = m * 16 + r + halfSel * 8;
            if (M < MROWS) {
                int t = M >> 1, p = M & 1;
                float wgt = time_w[(size_t)t * DDIM + d] * pol_w[(size_t)p * DDIM + d];
                partial += acc[m][r] * wgt;
            }
        }
    }
    atomicAdd(&gesture[d], partial);
}

__global__ void zero_ws_kernel(float* __restrict__ p, int n) {
    int i = blockIdx.x * 256 + threadIdx.x;
    if (i < n) p[i] = 0.f;
}

__global__ void sign_kernel(const float* __restrict__ g, float* __restrict__ out, int n) {
    int i = blockIdx.x * 256 + threadIdx.x;
    if (i < n) {
        float x = g[i];
        out[i] = (x > 0.f) ? 1.f : ((x < 0.f) ? -1.f : 0.f);
    }
}

extern "C" void kernel_launch(void* const* d_in, const int* in_sizes, int n_in,
                              void* d_out, int out_size, void* d_ws, size_t ws_size,
                              hipStream_t stream) {
    const float* hist   = (const float*)d_in[0];  // [100,2,128,128]
    const float* time_w = (const float*)d_in[1];  // [100,8192]
    const float* pol_w  = (const float*)d_in[2];  // [2,8192]
    const float* pos_w  = (const float*)d_in[3];  // [16384,8192]
    float* out     = (float*)d_out;               // [8192]
    float* gesture = (float*)d_ws;                // 8192 floats scratch

    (void)in_sizes; (void)n_in; (void)out_size; (void)ws_size;

    zero_ws_kernel<<<DDIM / 256, 256, 0, stream>>>(gesture, DDIM);

    dim3 grid(DDIM / BN, KSPLIT);   // 64 x 4 = 256 workgroups
    hv_gemm_bind_kernel<<<grid, 256, 0, stream>>>(hist, time_w, pol_w, pos_w, gesture);

    sign_kernel<<<DDIM / 256, 256, 0, stream>>>(gesture, out, DDIM);
}
